// Block_86595130622141
// MI455X (gfx1250) — compile-verified, matrix operand order checked by reading, not run on hardware
//
#include <hip/hip_runtime.h>
#include <hip/hip_bf16.h>

// ---------------------------------------------------------------------------
// Transformer block for MI455X (gfx1250), wave32 + WMMA f16 (fp32 accum).
// B=1, T=4096, C=1024, H=16, hd=64.
// GEMMs: double-buffered async global->LDS staging (ASYNCcnt) + WMMA from LDS,
// 128x128 block tile, 32x64 wave tile (8 WMMAs per K-step per wave).
// ---------------------------------------------------------------------------

#define T_SEQ 4096
#define C_EMB 1024
#define HD    64
#define NH    16
#define FF    (4 * C_EMB)
#define LDSW  40   // padded LDS row stride in halves (80B: 16B-aligned, conflict-free)

typedef __attribute__((ext_vector_type(16))) _Float16 v16h;
typedef __attribute__((ext_vector_type(8)))  _Float16 v8h;
typedef __attribute__((ext_vector_type(8)))  float    v8f;

union F16Frag { v16h v; v8h h[2]; };

// Build a 16-half WMMA operand fragment from two contiguous 16B runs.
// A-layout (16x32 f16): lane l holds row m=l&15; elems e -> K = e+(e&8)+8*(l>>4),
// i.e. halves [8h, 8h+8) and [16+8h, 24+8h). B mirrors A with lane = N column.
__device__ __forceinline__ v16h frag2(const _Float16* p0, const _Float16* p1) {
  F16Frag u;
  u.h[0] = *(const v8h*)p0;
  u.h[1] = *(const v8h*)p1;
  return u.v;
}

__device__ __forceinline__ v8f wmma_f16(v16h a, v16h b, v8f c) {
  return __builtin_amdgcn_wmma_f32_16x16x32_f16(
      /*neg_a=*/false, a, /*neg_b=*/false, b,
      /*c_mod=*/(short)0, c, /*reuse_a=*/false, /*reuse_b=*/false);
}

// Async DMA: 16B global -> LDS, tracked by ASYNCcnt (gfx1250).
// LDS byte address = low 32 bits of the generic pointer (flat->LDS truncation).
__device__ __forceinline__ void async_ld16(void* lds_ptr, const _Float16* gptr) {
  uint32_t l = (uint32_t)(uintptr_t)lds_ptr;
  asm volatile("global_load_async_to_lds_b128 %0, %1, off"
               :: "v"(l), "v"((unsigned long long)(uintptr_t)gptr) : "memory");
}
__device__ __forceinline__ void wait_async0() {
  asm volatile("s_wait_asynccnt 0" ::: "memory");
}

// ---------------------------------------------------------------------------
// Weight prep: fp32 [batch][R][Cc] -> f16 transposed [batch][Cc][R]
// (B operands are stored N-major so fragments load as 2x b128.)
// ---------------------------------------------------------------------------
__global__ __launch_bounds__(256) void tcvt_kernel(
    const float* __restrict__ src, _Float16* __restrict__ dst, int R, int Cc) {
  size_t base = (size_t)blockIdx.z * R * Cc;
  int idx = blockIdx.x * 256 + threadIdx.x;
  if (idx < R * Cc) {
    int r = idx / Cc, c = idx % Cc;
    dst[base + (size_t)c * R + r] = (_Float16)src[base + idx];
  }
}

// ---------------------------------------------------------------------------
// LayerNorm: one block per token row, fp32 in -> f16 out.
// ---------------------------------------------------------------------------
__global__ __launch_bounds__(256) void ln_kernel(
    const float* __restrict__ x, const float* __restrict__ g,
    const float* __restrict__ be, _Float16* __restrict__ out) {
  const int t = blockIdx.x;
  const int tid = threadIdx.x;
  const float* row = x + (size_t)t * C_EMB;
  __shared__ float rs[256], rq[256];
  float s = 0.f, sq = 0.f;
  for (int c = tid; c < C_EMB; c += 256) {
    float v = row[c];
    s += v; sq += v * v;
  }
  rs[tid] = s; rq[tid] = sq;
  __syncthreads();
  for (int off = 128; off > 0; off >>= 1) {
    if (tid < off) { rs[tid] += rs[tid + off]; rq[tid] += rq[tid + off]; }
    __syncthreads();
  }
  const float mu  = rs[0] * (1.f / C_EMB);
  const float var = rq[0] * (1.f / C_EMB) - mu * mu;
  const float inv = rsqrtf(var + 1e-5f);
  for (int c = tid; c < C_EMB; c += 256)
    out[(size_t)t * C_EMB + c] = (_Float16)((row[c] - mu) * inv * g[c] + be[c]);
}

// ---------------------------------------------------------------------------
// WMMA GEMM with double-buffered async->LDS staging:
//   out[M,N] = A[M,K](f16 row-major) * Bt[N,K](f16 N-major)
// 256 threads = 8 waves; block tile 128(M) x 128(N); wave tile 32 x 64
// (2 A-frags x 4 B-frags -> 8 WMMAs per K-step, loads hoisted ahead of MACs).
// Per K-step (32): stage A 128x32 + B 128x32 via global_load_async_to_lds_b128
// (4 x 16B per thread) while computing from the other buffer.
// Epilogue: optional bias / relu / fp32 residual; f16 row / f16 transposed /
// fp32 stores. M%128==0, N%128==0, K%32==0 (true for all calls here).
// ---------------------------------------------------------------------------
__global__ __launch_bounds__(256) void gemm_wmma(
    const _Float16* __restrict__ A, const _Float16* __restrict__ Bt,
    int M, int N, int K,
    const float* __restrict__ bias, const float* __restrict__ resid,
    float* __restrict__ outF, _Float16* __restrict__ outH,
    int transOut, int relu) {
  const int tid  = threadIdx.x;
  const int lane = tid & 31;
  const int w    = tid >> 5;
  const int nl   = lane & 15;
  const int hh   = lane >> 4;
  const int bm   = blockIdx.x * 128;
  const int bn   = blockIdx.y * 128;

  __shared__ __align__(16) _Float16 sA[2][128][LDSW];
  __shared__ __align__(16) _Float16 sB[2][128][LDSW];

  // staging: thread -> (row, 16B segment); A: 512 xfers, B: 512 (2+2 each).
  const int lrow = tid >> 2;       // 0..63
  const int lseg = (tid & 3) * 8;  // halves
  const _Float16* gA0 = A  + (size_t)(bm + lrow) * K + lseg;
  const _Float16* gA1 = A  + (size_t)(bm + 64 + lrow) * K + lseg;
  const _Float16* gB0 = Bt + (size_t)(bn + lrow) * K + lseg;
  const _Float16* gB1 = Bt + (size_t)(bn + 64 + lrow) * K + lseg;

  v8f z = {};
  v8f acc[2][4];
#pragma unroll
  for (int ri = 0; ri < 2; ++ri)
#pragma unroll
    for (int j = 0; j < 4; ++j) acc[ri][j] = z;

  const int o0  = 8 * hh, o1 = 16 + 8 * hh;
  const int tma = (w & 3) * 32 + nl;  // A row (first 16-row group) for this lane
  const int tnb = (w >> 2) * 64;      // B row base within tile for this wave

  // prologue: stage k0 = 0 into buffer 0
  async_ld16(&sA[0][lrow][lseg], gA0);
  async_ld16(&sA[0][64 + lrow][lseg], gA1);
  async_ld16(&sB[0][lrow][lseg], gB0);
  async_ld16(&sB[0][64 + lrow][lseg], gB1);

  int buf = 0;
  for (int k0 = 0; k0 < K; k0 += 32) {
    wait_async0();       // own DMA landed
    __syncthreads();     // everyone's DMA landed; prev buffer fully consumed
    if (k0 + 32 < K) {
      const int nb = buf ^ 1;
      async_ld16(&sA[nb][lrow][lseg], gA0 + k0 + 32);
      async_ld16(&sA[nb][64 + lrow][lseg], gA1 + k0 + 32);
      async_ld16(&sB[nb][lrow][lseg], gB0 + k0 + 32);
      async_ld16(&sB[nb][64 + lrow][lseg], gB1 + k0 + 32);
    }
    // hoist all fragment loads (12 x ds_load_b128) ahead of the 8 WMMAs
    const _Float16* a0 = &sA[buf][tma][0];
    const _Float16* a1 = a0 + 16 * LDSW;
    v16h af0 = frag2(a0 + o0, a0 + o1);
    v16h af1 = frag2(a1 + o0, a1 + o1);
    const _Float16* b0 = &sB[buf][tnb +  0 + nl][0];
    const _Float16* b1 = &sB[buf][tnb + 16 + nl][0];
    const _Float16* b2 = &sB[buf][tnb + 32 + nl][0];
    const _Float16* b3 = &sB[buf][tnb + 48 + nl][0];
    v16h bf0 = frag2(b0 + o0, b0 + o1);
    v16h bf1 = frag2(b1 + o0, b1 + o1);
    v16h bf2 = frag2(b2 + o0, b2 + o1);
    v16h bf3 = frag2(b3 + o0, b3 + o1);
    acc[0][0] = wmma_f16(af0, bf0, acc[0][0]);
    acc[0][1] = wmma_f16(af0, bf1, acc[0][1]);
    acc[0][2] = wmma_f16(af0, bf2, acc[0][2]);
    acc[0][3] = wmma_f16(af0, bf3, acc[0][3]);
    acc[1][0] = wmma_f16(af1, bf0, acc[1][0]);
    acc[1][1] = wmma_f16(af1, bf1, acc[1][1]);
    acc[1][2] = wmma_f16(af1, bf2, acc[1][2]);
    acc[1][3] = wmma_f16(af1, bf3, acc[1][3]);
    buf ^= 1;
  }

  // C/D layout: vgpr r, lane -> M = tm + 16*ri + r + 8*hh, N = tn + 16*j + nl
  const int tm = bm + (w & 3) * 32;
  const int tn = bn + (w >> 2) * 64;
#pragma unroll
  for (int ri = 0; ri < 2; ++ri) {
#pragma unroll
    for (int j = 0; j < 4; ++j) {
      const int n = tn + j * 16 + nl;
      const float bv = bias ? bias[n] : 0.f;
#pragma unroll
      for (int r = 0; r < 8; ++r) {
        const int m = tm + ri * 16 + r + 8 * hh;
        float val = acc[ri][j][r] + bv;
        if (relu) val = fmaxf(val, 0.f);
        if (resid) val += resid[(size_t)m * N + n];
        if (outF) {
          outF[(size_t)m * N + n] = val;
        } else if (transOut) {
          outH[(size_t)n * M + m] = (_Float16)val;   // transposed store (V^T)
        } else {
          outH[(size_t)m * N + n] = (_Float16)val;
        }
      }
    }
  }
}

// ---------------------------------------------------------------------------
// Flash attention (causal, online softmax). One wave = 16 query rows of one
// head; key tiles of 32. q,k: f16 [T][C] (col = h*64+d). vt: f16 [C][T].
// o: f16 [T][C]. scale = C^-0.5 = 1/32 (matches the reference!).
// Wave-divergent causal loop -> per-wave LDS staging only (no block barrier).
// ---------------------------------------------------------------------------
__global__ __launch_bounds__(256) void attn_wmma(
    const _Float16* __restrict__ q, const _Float16* __restrict__ k,
    const _Float16* __restrict__ vt, _Float16* __restrict__ o) {
  const int lane = threadIdx.x & 31;
  const int w    = threadIdx.x >> 5;
  const int nl   = lane & 15;
  const int hh   = lane >> 4;
  const int h    = blockIdx.y;
  const int qb   = (blockIdx.x * 8 + w) * 16;

  // per-wave staging: D-layout P -> A-layout P (row stride 40 keeps 16B align)
  __shared__ __align__(16) _Float16 sP[8][16][LDSW];

  const _Float16* qrow = q + (size_t)(qb + nl) * C_EMB + h * HD;
  const int o0 = 8 * hh, o1 = 16 + 8 * hh;

  v8f z = {};
  v8f acc[4]; acc[0] = z; acc[1] = z; acc[2] = z; acc[3] = z;
  float mrow[8], ssum[8];
#pragma unroll
  for (int r = 0; r < 8; ++r) { mrow[r] = -1e30f; ssum[r] = 0.f; }
  const float scale = 0.03125f;  // 1024^-0.5

  for (int kb = 0; kb <= qb + 15; kb += 32) {
    if (kb + 32 <= qb + 15) {  // prefetch next key tile (k row + v^t row)
      __builtin_prefetch(k + (size_t)(kb + 32 + nl) * C_EMB + h * HD, 0, 0);
      __builtin_prefetch(vt + (size_t)(h * HD + nl) * T_SEQ + kb + 32, 0, 0);
    }
    // S = Q * K^T for keys [kb, kb+32): two 16x16 D tiles, K-dim = hd = 64
    v8f s0 = z, s1 = z;
#pragma unroll
    for (int st = 0; st < 2; ++st) {
      const int ko = st * 32;
      v16h af = frag2(qrow + ko + o0, qrow + ko + o1);
      const _Float16* k0r = k + (size_t)(kb + nl) * C_EMB + h * HD + ko;
      const _Float16* k1r = k0r + (size_t)16 * C_EMB;
      v16h bf0 = frag2(k0r + o0, k0r + o1);
      v16h bf1 = frag2(k1r + o0, k1r + o1);
      s0 = wmma_f16(af, bf0, s0);
      s1 = wmma_f16(af, bf1, s1);
    }

    // scale + causal mask; row = qb + r + 8*hh, key = kb + (16t) + nl
    float a8[8], b8[8], t8[8];
#pragma unroll
    for (int r = 0; r < 8; ++r) {
      const int rq = qb + r + 8 * hh;
      float a = s0[r] * scale; if (kb + nl      > rq) a = -1e30f;
      float b = s1[r] * scale; if (kb + 16 + nl > rq) b = -1e30f;
      a8[r] = a; b8[r] = b; t8[r] = fmaxf(a, b);
    }
    // row max across the 16-lane half-group (rows never cross halves)
#pragma unroll
    for (int mk = 1; mk < 16; mk <<= 1)
#pragma unroll
      for (int r = 0; r < 8; ++r)
        t8[r] = fmaxf(t8[r], __shfl_xor(t8[r], mk, 32));

    float p8[8];
#pragma unroll
    for (int r = 0; r < 8; ++r) {
      const float mn    = fmaxf(mrow[r], t8[r]);
      const float alpha = __expf(mrow[r] - mn);
      mrow[r] = mn;
      const float p0 = __expf(a8[r] - mn);
      const float p1 = __expf(b8[r] - mn);
      p8[r] = p0 + p1;
      ssum[r] *= alpha;
      acc[0][r] *= alpha; acc[1][r] *= alpha;
      acc[2][r] *= alpha; acc[3][r] *= alpha;
      const int m = r + 8 * hh;
      sP[w][m][nl]      = (_Float16)p0;
      sP[w][m][16 + nl] = (_Float16)p1;
    }
#pragma unroll
    for (int mk = 1; mk < 16; mk <<= 1)
#pragma unroll
      for (int r = 0; r < 8; ++r)
        p8[r] += __shfl_xor(p8[r], mk, 32);
#pragma unroll
    for (int r = 0; r < 8; ++r) ssum[r] += p8[r];

    // wave-private LDS: order stores before re-reading in A-layout
    asm volatile("s_wait_dscnt 0" ::: "memory");

    v16h pf = frag2(&sP[w][nl][o0], &sP[w][nl][o1]);
    // O += P * V   (K = 32 keys, N = 64 = hd via 4 tiles), V^T rows contiguous
#pragma unroll
    for (int j = 0; j < 4; ++j) {
      const _Float16* vr = vt + (size_t)(h * HD + j * 16 + nl) * T_SEQ + kb;
      v16h bf = frag2(vr + o0, vr + o1);
      acc[j] = wmma_f16(pf, bf, acc[j]);
    }
  }

  // normalize + store f16 [T][C], col = h*64 + 16j + nl
#pragma unroll
  for (int r = 0; r < 8; ++r) {
    const float inv = 1.f / ssum[r];
    const size_t row = (size_t)(qb + r + 8 * hh) * C_EMB + h * HD;
#pragma unroll
    for (int j = 0; j < 4; ++j)
      o[row + j * 16 + nl] = (_Float16)(acc[j][r] * inv);
  }
}

// ---------------------------------------------------------------------------
// Launch
// ---------------------------------------------------------------------------
extern "C" void kernel_launch(void* const* d_in, const int* in_sizes, int n_in,
                              void* d_out, int out_size, void* d_ws, size_t ws_size,
                              hipStream_t stream) {
  const float* x  = (const float*)d_in[0];
  const float* Wq = (const float*)d_in[1];
  const float* Wk = (const float*)d_in[2];
  const float* Wv = (const float*)d_in[3];
  const float* Wp = (const float*)d_in[4];
  const float* bp = (const float*)d_in[5];
  const float* W1 = (const float*)d_in[6];
  const float* b1 = (const float*)d_in[7];
  const float* W2 = (const float*)d_in[8];
  const float* b2 = (const float*)d_in[9];
  const float* g1 = (const float*)d_in[10];
  const float* be1 = (const float*)d_in[11];
  const float* g2 = (const float*)d_in[12];
  const float* be2 = (const float*)d_in[13];
  float* out = (float*)d_out;

  char* p = (char*)d_ws;
  auto alloc = [&](size_t bytes) -> char* {
    char* r = p;
    p += (bytes + 255) & ~(size_t)255;
    return r;
  };
  _Float16* wq_t = (_Float16*)alloc((size_t)C_EMB * C_EMB * 2);
  _Float16* wk_t = (_Float16*)alloc((size_t)C_EMB * C_EMB * 2);
  _Float16* wv_t = (_Float16*)alloc((size_t)C_EMB * C_EMB * 2);
  _Float16* wp_t = (_Float16*)alloc((size_t)C_EMB * C_EMB * 2);
  _Float16* w1_t = (_Float16*)alloc((size_t)C_EMB * FF * 2);
  _Float16* w2_t = (_Float16*)alloc((size_t)FF * C_EMB * 2);
  _Float16* h1   = (_Float16*)alloc((size_t)T_SEQ * C_EMB * 2);
  _Float16* qm   = (_Float16*)alloc((size_t)T_SEQ * C_EMB * 2);
  _Float16* km   = (_Float16*)alloc((size_t)T_SEQ * C_EMB * 2);
  _Float16* vtm  = (_Float16*)alloc((size_t)C_EMB * T_SEQ * 2);
  _Float16* am   = (_Float16*)alloc((size_t)T_SEQ * C_EMB * 2);
  float*    x1   = (float*)   alloc((size_t)T_SEQ * C_EMB * 4);
  _Float16* h2   = (_Float16*)alloc((size_t)T_SEQ * C_EMB * 2);
  _Float16* ff1  = (_Float16*)alloc((size_t)T_SEQ * FF * 2);

  const dim3 blk(256);

  // --- weight conversion (transpose to N-major f16) ---
  // Wq/Wk/Wv: [H][C][hd] -> [h*64+d][C]   (batch = H)
  tcvt_kernel<<<dim3((C_EMB * HD + 255) / 256, 1, NH), blk, 0, stream>>>(Wq, wq_t, C_EMB, HD);
  tcvt_kernel<<<dim3((C_EMB * HD + 255) / 256, 1, NH), blk, 0, stream>>>(Wk, wk_t, C_EMB, HD);
  tcvt_kernel<<<dim3((C_EMB * HD + 255) / 256, 1, NH), blk, 0, stream>>>(Wv, wv_t, C_EMB, HD);
  tcvt_kernel<<<dim3((C_EMB * C_EMB + 255) / 256, 1, 1), blk, 0, stream>>>(Wp, wp_t, C_EMB, C_EMB);
  tcvt_kernel<<<dim3((C_EMB * FF + 255) / 256, 1, 1), blk, 0, stream>>>(W1, w1_t, C_EMB, FF);
  tcvt_kernel<<<dim3((FF * C_EMB + 255) / 256, 1, 1), blk, 0, stream>>>(W2, w2_t, FF, C_EMB);

  // --- LN1 ---
  ln_kernel<<<dim3(T_SEQ), blk, 0, stream>>>(x, g1, be1, h1);

  // --- QKV projections (heads concatenated along N) ---
  gemm_wmma<<<dim3(T_SEQ / 128, C_EMB / 128), blk, 0, stream>>>(
      h1, wq_t, T_SEQ, C_EMB, C_EMB, nullptr, nullptr, nullptr, qm, 0, 0);
  gemm_wmma<<<dim3(T_SEQ / 128, C_EMB / 128), blk, 0, stream>>>(
      h1, wk_t, T_SEQ, C_EMB, C_EMB, nullptr, nullptr, nullptr, km, 0, 0);
  gemm_wmma<<<dim3(T_SEQ / 128, C_EMB / 128), blk, 0, stream>>>(
      h1, wv_t, T_SEQ, C_EMB, C_EMB, nullptr, nullptr, nullptr, vtm, 1, 0);  // V^T

  // --- causal flash attention ---
  attn_wmma<<<dim3(T_SEQ / 128, NH), blk, 0, stream>>>(qm, km, vtm, am);

  // --- output projection + bias + residual:  x1 = x + attn*Wp + bp ---
  gemm_wmma<<<dim3(T_SEQ / 128, C_EMB / 128), blk, 0, stream>>>(
      am, wp_t, T_SEQ, C_EMB, C_EMB, bp, x, x1, nullptr, 0, 0);

  // --- LN2 ---
  ln_kernel<<<dim3(T_SEQ), blk, 0, stream>>>(x1, g2, be2, h2);

  // --- FFN:  ff1 = relu(h2*W1 + b1);  out = x1 + ff1*W2 + b2 ---
  gemm_wmma<<<dim3(T_SEQ / 128, FF / 128), blk, 0, stream>>>(
      h2, w1_t, T_SEQ, FF, C_EMB, b1, nullptr, nullptr, ff1, 0, 1);
  gemm_wmma<<<dim3(T_SEQ / 128, C_EMB / 128), blk, 0, stream>>>(
      ff1, w2_t, T_SEQ, C_EMB, FF, b2, x1, out, nullptr, 0, 0);
}
